// AngularMultiCenterEmotionBall_16776142258425
// MI455X (gfx1250) — matrix operand order repaired and checked by reading, not compiled, not run on hardware
//
#include <hip/hip_runtime.h>
#include <hip/hip_bf16.h>

#define B_N    16384
#define ZD_N   1024
#define C_N    7
#define K_N    4
#define DSH_N  768
#define DSP_N  256
#define TAU_F  0.15f

typedef __attribute__((ext_vector_type(16))) __bf16          v16bf;
typedef __attribute__((ext_vector_type(8)))  float           v8f;
typedef __attribute__((ext_vector_type(8)))  unsigned short  us8;
typedef __attribute__((ext_vector_type(4)))  unsigned short  us4;

union FragBF {
    us8            h[2];
    unsigned short u[16];
    v16bf          v;
};

__device__ __forceinline__ float bf2f(unsigned short h) {
    return __uint_as_float(((unsigned)h) << 16);
}
__device__ __forceinline__ unsigned short f2bf(float f) {
    unsigned u = __float_as_uint(f);
    unsigned r = u + 0x7FFFu + ((u >> 16) & 1u);   // round-to-nearest-even
    return (unsigned short)(r >> 16);
}

// ---------------------------------------------------------------- init -----
__global__ void k_init(float* acc) {
    if (threadIdx.x < 46) acc[threadIdx.x] = 0.0f;  // 4 accums + 28 sum_q + 7 counts + 7 hcond
}

// ----------------------------------------------------- LayerNorm -> bf16 ---
__global__ void k_layernorm(const float* __restrict__ z,
                            const float* __restrict__ gamma,
                            const float* __restrict__ beta,
                            unsigned short* __restrict__ z0) {
    const int row = blockIdx.x;
    const float* zr = z + (size_t)row * ZD_N;
    float x[4], s = 0.f, ss = 0.f;
    for (int i = 0; i < 4; ++i) {
        x[i] = zr[threadIdx.x + i * 256];
        s += x[i]; ss += x[i] * x[i];
    }
    __shared__ float r0[8], r1[8];
    for (int o = 16; o > 0; o >>= 1) { s += __shfl_down(s, o); ss += __shfl_down(ss, o); }
    const int wid = threadIdx.x >> 5, lane = threadIdx.x & 31;
    if (lane == 0) { r0[wid] = s; r1[wid] = ss; }
    __syncthreads();
    if (wid == 0) {
        s  = (lane < 8) ? r0[lane] : 0.f;
        ss = (lane < 8) ? r1[lane] : 0.f;
        for (int o = 4; o > 0; o >>= 1) { s += __shfl_down(s, o); ss += __shfl_down(ss, o); }
        if (lane == 0) { r0[0] = s; r1[0] = ss; }
    }
    __syncthreads();
    s = r0[0]; ss = r1[0];
    const float mu  = s * (1.0f / ZD_N);
    const float var = ss * (1.0f / ZD_N) - mu * mu;
    const float inv = rsqrtf(var + 1e-5f);
    unsigned short* o = z0 + (size_t)row * ZD_N;
    for (int i = 0; i < 4; ++i) {
        const int c = threadIdx.x + i * 256;
        o[c] = f2bf((x[i] - mu) * inv * gamma[c] + beta[c]);
    }
}

// ------------------------------------------- W [K][N] -> W^T [N][K] bf16 ---
__global__ void k_wt_bf16(const float* __restrict__ W, unsigned short* __restrict__ WT, int N) {
    const int idx = blockIdx.x * 256 + threadIdx.x;   // over N*1024
    const int n = idx >> 10, k = idx & 1023;
    if (n < N) WT[idx] = f2bf(W[(size_t)k * N + n]);
}

// --------------------------- WMMA GEMM: out[b][n] = z0[b][:] . WT[n][:] ----
// One 16x16 output tile per wave; K-loop of 32 with v_wmma_f32_16x16x32_bf16.
__global__ void k_gemm_bias(const unsigned short* __restrict__ A,   // [B][1024] bf16
                            const unsigned short* __restrict__ WT,  // [N][1024] bf16
                            const float* __restrict__ bias,
                            unsigned short* __restrict__ out,       // [B][N] bf16
                            int N) {
    const int wave = threadIdx.x >> 5, lane = threadIdx.x & 31;
    const int hi = lane >> 4, lm = lane & 15;
    const int row0 = blockIdx.x * 16;
    const int n0   = blockIdx.y * 128 + wave * 16;
    const unsigned short* arow = A  + (size_t)(row0 + lm) * ZD_N;
    const unsigned short* brow = WT + (size_t)(n0   + lm) * ZD_N;
    v8f acc = {};
    for (int kc = 0; kc < ZD_N; kc += 32) {
        __builtin_prefetch(arow + kc + 256, 0, 1);
        __builtin_prefetch(brow + kc + 256, 0, 1);
        FragBF fa, fb;
        // A 16x32: elems 0..7 -> K = kc+hi*8 .. +7 ; elems 8..15 -> K = kc+16+hi*8 ..
        fa.h[0] = *(const us8*)(arow + kc + hi * 8);
        fa.h[1] = *(const us8*)(arow + kc + 16 + hi * 8);
        // B 32x16: elems 0..7 -> K = kc+hi*8 .. +7 ; elems 8..15 -> K = kc+16+hi*8 ..
        fb.h[0] = *(const us8*)(brow + kc + hi * 8);
        fb.h[1] = *(const us8*)(brow + kc + 16 + hi * 8);
        acc = __builtin_amdgcn_wmma_f32_16x16x32_bf16(false, fa.v, false, fb.v,
                                                      (short)0, acc, false, false);
    }
    const float bv = bias[n0 + lm];
    for (int r = 0; r < 8; ++r) {
        const int m = r + hi * 8;                      // D: VGPR r -> row r (+8 for hi lanes)
        out[(size_t)(row0 + m) * N + n0 + lm] = f2bf(acc[r] + bv);
    }
}

// ------------------- center normalize + overlap/diversity (one block) ------
__global__ void k_centers(const float* __restrict__ centers,
                          float* __restrict__ cnorm,
                          float* __restrict__ acc) {
    const int wid = threadIdx.x >> 5, lane = threadIdx.x & 31;
    for (int r = wid; r < C_N * K_N; r += 8) {
        const float* cr = centers + r * DSH_N;
        float ss = 0.f;
        for (int j = lane; j < DSH_N; j += 32) { const float v = cr[j]; ss += v * v; }
        for (int o = 16; o > 0; o >>= 1) ss += __shfl_down(ss, o);
        ss = __shfl(ss, 0);
        const float inv = 1.f / fmaxf(sqrtf(ss), 1e-12f);
        for (int j = lane; j < DSH_N; j += 32) cnorm[r * DSH_N + j] = cr[j] * inv;
    }
    __syncthreads();
    float ov = 0.f, dv = 0.f;
    int pidx = 0;
    for (int i = 0; i < C_N * K_N; ++i)
        for (int j = i + 1; j < C_N * K_N; ++j) {
            if (((pidx++) & 7) != wid) continue;       // wave-uniform distribution
            float d = 0.f;
            for (int t = lane; t < DSH_N; t += 32) d += cnorm[i * DSH_N + t] * cnorm[j * DSH_N + t];
            for (int o = 16; o > 0; o >>= 1) d += __shfl_down(d, o);
            if (lane == 0) {
                if (i / K_N == j / K_N) dv += fmaxf(d - 0.8f, 0.f);
                else                    ov += 2.f * fmaxf(d - 0.3f, 0.f);   // symmetric pairs
            }
        }
    if (lane == 0) { atomicAdd(&acc[1], ov); atomicAdd(&acc[2], dv); }
}

// ---------------- per-sample: sim, softmax q, intra + balance stats --------
__global__ void k_persample(const unsigned short* __restrict__ zsh,
                            const int* __restrict__ labels,
                            const float* __restrict__ cnorm,
                            const float* __restrict__ radii,
                            float* __restrict__ acc,      // [0]=intra
                            float* __restrict__ sum_q,    // [7][4]
                            float* __restrict__ counts,   // [7]
                            float* __restrict__ hcond) {  // [7]
    const int wid = threadIdx.x >> 5, lane = threadIdx.x & 31;
    const int b = blockIdx.x * 8 + wid;
    const unsigned short* zr = zsh + (size_t)b * DSH_N;
    const int lab = labels[b];
    const float* c0 = cnorm + (size_t)lab * K_N * DSH_N;
    float d0 = 0, d1 = 0, d2 = 0, d3 = 0, ss = 0;
    for (int j = lane; j < DSH_N; j += 32) {
        const float x = bf2f(zr[j]);
        ss += x * x;
        d0 += x * c0[j];
        d1 += x * c0[DSH_N + j];
        d2 += x * c0[2 * DSH_N + j];
        d3 += x * c0[3 * DSH_N + j];
    }
    for (int o = 16; o > 0; o >>= 1) {
        ss += __shfl_down(ss, o); d0 += __shfl_down(d0, o); d1 += __shfl_down(d1, o);
        d2 += __shfl_down(d2, o); d3 += __shfl_down(d3, o);
    }
    if (lane == 0) {
        const float inv = 1.f / fmaxf(sqrtf(ss), 1e-12f);
        float s[4] = { d0 * inv, d1 * inv, d2 * inv, d3 * inv };
        const float mx = fmaxf(fmaxf(s[0], s[1]), fmaxf(s[2], s[3]));
        float e[4], te = 0.f;
        for (int k = 0; k < 4; ++k) { e[k] = __expf((s[k] - mx) * (1.0f / TAU_F)); te += e[k]; }
        float dist_w = 0.f, r_w = 0.f, Hc = 0.f;
        for (int k = 0; k < 4; ++k) {
            const float q = e[k] / te;
            dist_w += q * (1.f - s[k]);
            r_w    += q * radii[lab * K_N + k];
            Hc     -= q * __logf(q + 1e-8f);
            atomicAdd(&sum_q[lab * K_N + k], q);
        }
        atomicAdd(&acc[0], fmaxf(dist_w - r_w, 0.f));
        atomicAdd(&counts[lab], 1.f);
        atomicAdd(&hcond[lab], Hc);
    }
}

// ---------------- column stats (for ortho norms + variance floor) ----------
__global__ void k_colstats(const unsigned short* __restrict__ zsh,
                           const unsigned short* __restrict__ zsp,
                           float* __restrict__ ss_sh, float* __restrict__ ss_sp,
                           float* __restrict__ sum_sp) {
    const int col = blockIdx.x * 256 + threadIdx.x;   // 0..1023
    if (col < DSH_N) {
        float ss = 0.f;
        for (int b = 0; b < B_N; ++b) { const float x = bf2f(zsh[(size_t)b * DSH_N + col]); ss += x * x; }
        ss_sh[col] = ss;
    } else {
        const int c = col - DSH_N;                    // 0..255
        float ss = 0.f, s = 0.f;
        for (int b = 0; b < B_N; ++b) { const float x = bf2f(zsp[(size_t)b * DSP_N + c]); ss += x * x; s += x; }
        ss_sp[c] = ss; sum_sp[c] = s;
    }
}

// ---------------- ortho: corr = z_sh^T @ z_sp via WMMA w/ LDS transpose ----
__global__ void k_ortho(const unsigned short* __restrict__ zsh,
                        const unsigned short* __restrict__ zsp,
                        const float* __restrict__ ss_sh,
                        const float* __restrict__ ss_sp,
                        float* __restrict__ acc) {
    __shared__ unsigned short lA[32 * 16];   // [k][m]  (staged transpose of z_sh tile)
    __shared__ unsigned short lB[32 * 64];   // [k][n]
    const int wave = threadIdx.x >> 5, lane = threadIdx.x & 31;
    const int hi = lane >> 4, lm = lane & 15;
    const int m0 = blockIdx.x * 16;          // z_sh column tile
    const int nb = blockIdx.y * 64;          // z_sp column block (4 waves x 16)
    v8f gacc = {};
    for (int kc = 0; kc < B_N; kc += 32) {
        const int row = threadIdx.x >> 2, part = threadIdx.x & 3;
        *(us4*)(lA + row * 16 + part * 4) =
            *(const us4*)(zsh + (size_t)(kc + row) * DSH_N + m0 + part * 4);
        *(us8*)(lB + row * 64 + part * 16) =
            *(const us8*)(zsp + (size_t)(kc + row) * DSP_N + nb + part * 16);
        *(us8*)(lB + row * 64 + part * 16 + 8) =
            *(const us8*)(zsp + (size_t)(kc + row) * DSP_N + nb + part * 16 + 8);
        __syncthreads();
        FragBF fa, fb;
        for (int e = 0; e < 16; ++e) {
            const int ka = (e & 7) + ((e & 8) ? 16 : 0) + hi * 8;   // A 16x32 layout
            fa.u[e] = lA[ka * 16 + lm];
            const int kb = (e & 7) + ((e & 8) ? 16 : 0) + hi * 8;   // B 32x16 layout
            fb.u[e] = lB[kb * 64 + wave * 16 + lm];
        }
        gacc = __builtin_amdgcn_wmma_f32_16x16x32_bf16(false, fa.v, false, fb.v,
                                                       (short)0, gacc, false, false);
        __syncthreads();
    }
    const float nsp = fmaxf(sqrtf(ss_sp[nb + wave * 16 + lm]), 1e-12f);
    float local = 0.f;
    for (int r = 0; r < 8; ++r) {
        const int m = m0 + r + hi * 8;
        const float nsh = fmaxf(sqrtf(ss_sh[m]), 1e-12f);
        const float corr = gacc[r] / (nsh * nsp);
        local += corr * corr;
    }
    for (int o = 16; o > 0; o >>= 1) local += __shfl_down(local, o);
    if (lane == 0) atomicAdd(&acc[3], local);
}

// ------------------------------------------------ final scalar combine -----
__global__ void k_final(const float* __restrict__ acc,
                        const float* __restrict__ sum_q,
                        const float* __restrict__ counts,
                        const float* __restrict__ hcond,
                        const float* __restrict__ ss_sp,
                        const float* __restrict__ sum_sp,
                        float* __restrict__ out) {
    __shared__ float red[8];
    const int lane = threadIdx.x & 31, wid = threadIdx.x >> 5;
    const int c = threadIdx.x;                 // 256 threads = 256 sp columns
    const float mean = sum_sp[c] * (1.0f / B_N);
    const float var  = ss_sp[c] * (1.0f / B_N) - mean * mean;
    float v = fmaxf(0.05f - var, 0.f);
    for (int o = 16; o > 0; o >>= 1) v += __shfl_down(v, o);
    if (lane == 0) red[wid] = v;
    __syncthreads();
    if (threadIdx.x == 0) {
        float vs = 0.f;
        for (int i = 0; i < 8; ++i) vs += red[i];
        const float L_var     = vs * (1.0f / DSP_N);
        const float L_intra   = acc[0] * (1.0f / B_N);
        const float L_overlap = acc[1] / (672.0f + 1e-6f);          // mask.sum() = 784-112
        const float L_div     = acc[2] * (1.0f / 42.0f);            // C*K*(K-1)/2
        const float L_ortho   = acc[3] * (1.0f / (DSH_N * DSP_N));
        float bal = 0.f; int valid = 0;
        for (int cc = 0; cc < C_N; ++cc) {
            const float cnt = counts[cc];
            if (cnt > 0.f) {
                float tot = 0.f;
                for (int k = 0; k < K_N; ++k) tot += sum_q[cc * K_N + k];
                float Hm = 0.f;
                for (int k = 0; k < K_N; ++k) {
                    const float p = sum_q[cc * K_N + k] / (tot + 1e-8f);
                    Hm -= p * __logf(p + 1e-8f);
                }
                const float Hc = hcond[cc] / fmaxf(cnt, 1.f);
                bal += __logf((float)K_N) - Hm + Hc;
                ++valid;
            }
        }
        const float L_balance = bal / (float)(valid > 0 ? valid : 1);
        const float L_ball = L_intra + 0.3f * L_overlap + 0.2f * L_div + 0.15f * L_balance;
        out[0] = L_ball + 0.02f * L_ortho + 0.005f * L_var;
    }
}

// ---------------------------------------------------------------------------
extern "C" void kernel_launch(void* const* d_in, const int* in_sizes, int n_in,
                              void* d_out, int out_size, void* d_ws, size_t ws_size,
                              hipStream_t stream) {
    const float* z       = (const float*)d_in[0];
    const int*   labels  = (const int*)  d_in[1];
    const float* gamma   = (const float*)d_in[2];
    const float* beta    = (const float*)d_in[3];
    const float* W_sh    = (const float*)d_in[4];
    const float* b_sh    = (const float*)d_in[5];
    const float* W_sp    = (const float*)d_in[6];
    const float* b_sp    = (const float*)d_in[7];
    const float* centers = (const float*)d_in[8];
    const float* radii   = (const float*)d_in[9];
    float* out = (float*)d_out;

    unsigned char* ws = (unsigned char*)d_ws;
    size_t off = 0;
    auto take = [&](size_t bytes) { size_t o = off; off = (off + bytes + 255) & ~(size_t)255; return o; };
    unsigned short* z0    = (unsigned short*)(ws + take((size_t)B_N * ZD_N * 2));
    unsigned short* wshT  = (unsigned short*)(ws + take((size_t)DSH_N * ZD_N * 2));
    unsigned short* wspT  = (unsigned short*)(ws + take((size_t)DSP_N * ZD_N * 2));
    unsigned short* zsh   = (unsigned short*)(ws + take((size_t)B_N * DSH_N * 2));
    unsigned short* zsp   = (unsigned short*)(ws + take((size_t)B_N * DSP_N * 2));
    float* cnorm  = (float*)(ws + take((size_t)C_N * K_N * DSH_N * 4));
    float* ss_sh  = (float*)(ws + take(DSH_N * 4));
    float* ss_sp  = (float*)(ws + take(DSP_N * 4));
    float* sum_sp = (float*)(ws + take(DSP_N * 4));
    float* accums = (float*)(ws + take(46 * 4));     // [0..3]=intra,ov,div,ortho
    float* sum_q  = accums + 4;                      // [7][4]
    float* counts = accums + 4 + 28;                 // [7]
    float* hcond  = accums + 4 + 28 + 7;             // [7]

    k_init<<<1, 64, 0, stream>>>(accums);
    k_layernorm<<<B_N, 256, 0, stream>>>(z, gamma, beta, z0);
    k_wt_bf16<<<(DSH_N * ZD_N) / 256, 256, 0, stream>>>(W_sh, wshT, DSH_N);
    k_wt_bf16<<<(DSP_N * ZD_N) / 256, 256, 0, stream>>>(W_sp, wspT, DSP_N);
    k_gemm_bias<<<dim3(B_N / 16, DSH_N / 128), 256, 0, stream>>>(z0, wshT, b_sh, zsh, DSH_N);
    k_gemm_bias<<<dim3(B_N / 16, DSP_N / 128), 256, 0, stream>>>(z0, wspT, b_sp, zsp, DSP_N);
    k_centers<<<1, 256, 0, stream>>>(centers, cnorm, accums);
    k_persample<<<B_N / 8, 256, 0, stream>>>(zsh, labels, cnorm, radii, accums, sum_q, counts, hcond);
    k_colstats<<<4, 256, 0, stream>>>(zsh, zsp, ss_sh, ss_sp, sum_sp);
    k_ortho<<<dim3(DSH_N / 16, DSP_N / 64), 128, 0, stream>>>(zsh, zsp, ss_sh, ss_sp, accums);
    k_final<<<1, 256, 0, stream>>>(accums, sum_q, counts, hcond, ss_sp, sum_sp, out);
}
